// VisionExperts_68977174774108
// MI455X (gfx1250) — compile-verified, matrix operand order checked by reading, not run on hardware
//
#include <hip/hip_runtime.h>
#include <hip/hip_bf16.h>

// ---------------------------------------------------------------------------
// MI455X (gfx1250) vision-MoE pipeline, bf16 WMMA (16x16x32, f32 accum),
// Tensor Data Mover (TDM) async tile staging when the builtin is available.
// Workspace requirement: ~298 MB (regions at bottom).
// ---------------------------------------------------------------------------

typedef __bf16 bf16_t;
typedef __attribute__((ext_vector_type(16))) __bf16 v16bf;
typedef __attribute__((ext_vector_type(8)))  __bf16 v8bf;
typedef __attribute__((ext_vector_type(8)))  float  v8f;

#define TB 256

#ifndef __has_builtin
#define __has_builtin(x) 0
#endif

#if defined(__HIP_DEVICE_COMPILE__) && \
    __has_builtin(__builtin_amdgcn_tensor_load_to_lds) && \
    __has_builtin(__builtin_amdgcn_s_wait_tensorcnt)
#define HAVE_TDM 1
#else
#define HAVE_TDM 0
#endif

// ROCm 7.2 (clang-22): 5-arg tensor_load_to_lds. amdgpu-toolchain (clang-23,
// therock-10.0 headers, which uniquely ship the TDM header): 6-arg form.
#if defined(__has_include)
#if __has_include(<hip/amd_detail/amd_gfx1250_TDM.h>)
#define TDM_SIX_ARGS 1
#else
#define TDM_SIX_ARGS 0
#endif
#else
#define TDM_SIX_ARGS 0
#endif

__device__ __forceinline__ int imin_dev(int a, int b) { return a < b ? a : b; }

#if HAVE_TDM
typedef unsigned int u32x4 __attribute__((ext_vector_type(4)));
typedef int          i32x4 __attribute__((ext_vector_type(4)));
typedef int          i32x8 __attribute__((ext_vector_type(8)));

// ---------------------------------------------------------------------------
// Issue one TDM DMA: 2-D tile [128 rows x 32 bf16] from row-major global
// (row stride = K elements) into LDS, with hardware padding of 4 dwords
// (16 B) after every 16 dwords (64 B = one 32-element row) => LDS row stride
// of 40 bf16 (bank-conflict-free for the 16-lane WMMA fragment reads).
// D# fields per CDNA5 ISA. All inputs wave-uniform.
// ---------------------------------------------------------------------------
__device__ __forceinline__ void tdm_load_tile(const bf16_t* gsrc, bf16_t* ldst, int K)
{
    unsigned long long ga = (unsigned long long)(const void*)gsrc;   // 57-bit
    unsigned lds_off = (unsigned)(unsigned long long)(void*)ldst;    // addr[31:0]
    // group0: count=1 | lds_addr | global_addr[56:0] | type=2 (bits 127:126)
    u32x4 g0 = { 1u, lds_off, (unsigned)ga,
                 (unsigned)(ga >> 32) | 0x80000000u };
    unsigned td0 = (unsigned)K;                 // tensor_dim0 (elements)
    // group1 d0: wg_mask=0 | data_size=1(2B)<<16 | pad_en<<20 |
    //            pad_interval=3(16dw)<<22 | pad_amount=3(4dw)<<25
    i32x8 g1 = {
        (int)0x06D10000u,
        (int)((td0 & 0xFFFFu) << 16),           // abar_addr=0 | td0.lo16
        (int)(td0 >> 16),                       // td0.hi16 | td1.lo16(=0)
        (int)(0x4000u | (32u << 16)),           // td1.hi16 (td1=1<<30) | tile_dim0=32
        (int)128,                               // tile_dim1=128 | tile_dim2=0
        (int)td0,                               // tensor_dim0_stride = K
        0, 0 };                                 // stride hi / dim1_stride = 0
    i32x4 gz = { 0, 0, 0, 0 };
#if TDM_SIX_ARGS
    i32x8 gz8 = { 0, 0, 0, 0, 0, 0, 0, 0 };
    __builtin_amdgcn_tensor_load_to_lds(g0, g1, gz, gz, gz8, 0);
#else
    __builtin_amdgcn_tensor_load_to_lds(g0, g1, gz, gz, 0);
#endif
}
#endif  // HAVE_TDM

// ---------------------------------------------------------------------------
// Patch packing: image [64,3,448,448] f32 -> A [rows, Kpad] bf16.
// Fuses align_corners bilinear input resize (448 -> S) when S != 448.
// Row = b*g*g + gy*g + gx ; column kk = c*p*p + py*p + px (reference layout).
// ---------------------------------------------------------------------------
__global__ void pack_patches_kernel(const float* __restrict__ x, bf16_t* __restrict__ A,
                                    int g, int p, int Kpad, int S, long total)
{
    long idx = (long)blockIdx.x * blockDim.x + threadIdx.x;
    if (idx >= total) return;
    int  kk  = (int)(idx % Kpad);
    long row = idx / Kpad;
    int  pp  = p * p;
    int  Kr  = 3 * pp;
    if (kk >= Kr) { A[idx] = (bf16_t)0.0f; return; }
    int  gx = (int)(row % g);
    long t  = row / g;
    int  gy = (int)(t % g);
    int  b  = (int)(t / g);
    int  c  = kk / pp;
    int  r2 = kk % pp;
    int  py = r2 / p, px = r2 % p;
    int  oy = gy * p + py, ox = gx * p + px;

    const float* img = x + ((long)b * 3 + c) * 448 * 448;
    float v;
    if (S == 448) {
        v = img[oy * 448 + ox];
    } else {
        float scale = 447.0f / (float)(S - 1);     // align_corners=True
        float fy = oy * scale, fx = ox * scale;
        int y0 = (int)fy, x0 = (int)fx;
        int y1 = imin_dev(y0 + 1, 447), x1 = imin_dev(x0 + 1, 447);
        float ty = fy - (float)y0, tx = fx - (float)x0;
        float a = img[y0 * 448 + x0], bq = img[y0 * 448 + x1];
        float cq = img[y1 * 448 + x0], dq = img[y1 * 448 + x1];
        v = (a * (1.0f - tx) + bq * tx) * (1.0f - ty) +
            (cq * (1.0f - tx) + dq * tx) * ty;
    }
    A[idx] = (bf16_t)v;
}

// ---------------------------------------------------------------------------
// Weight transpose+convert: W f32 [K,N] -> WT bf16 [N, Kpad], zero-padded K.
// ---------------------------------------------------------------------------
__global__ void convert_wT_kernel(const float* __restrict__ W, bf16_t* __restrict__ WT,
                                  int K, int N, int Kpad, long total)
{
    long idx = (long)blockIdx.x * blockDim.x + threadIdx.x;
    if (idx >= total) return;
    int  k = (int)(idx % Kpad);
    long n = idx / Kpad;
    WT[idx] = (bf16_t)((k < K) ? W[(long)k * N + n] : 0.0f);
}

// ---------------------------------------------------------------------------
// Feature-grid bilinear resize (align_corners): F bf16 [64, g*g, c] -> P bf16
// [64*1024, c] (32x32 output grid, token-major rows, channel = K for proj).
// ---------------------------------------------------------------------------
__global__ void resize_feat_kernel(const bf16_t* __restrict__ F, bf16_t* __restrict__ P,
                                   int g, int c, long total)
{
    long idx = (long)blockIdx.x * blockDim.x + threadIdx.x;
    if (idx >= total) return;
    int  ch = (int)(idx % c);
    long t  = idx / c;
    int  ox = (int)(t % 32);
    int  oy = (int)((t / 32) % 32);
    int  b  = (int)(t / 1024);
    float scale = (float)(g - 1) / 31.0f;
    float fy = oy * scale, fx = ox * scale;
    int y0 = (int)fy, x0 = (int)fx;
    int y1 = imin_dev(y0 + 1, g - 1), x1 = imin_dev(x0 + 1, g - 1);
    float ty = fy - (float)y0, tx = fx - (float)x0;
    const bf16_t* Fb = F + (long)b * g * g * c;
    float a  = (float)Fb[((long)(y0 * g + x0)) * c + ch];
    float bq = (float)Fb[((long)(y0 * g + x1)) * c + ch];
    float cq = (float)Fb[((long)(y1 * g + x0)) * c + ch];
    float dq = (float)Fb[((long)(y1 * g + x1)) * c + ch];
    float v = (a * (1.0f - tx) + bq * tx) * (1.0f - ty) +
              (cq * (1.0f - tx) + dq * tx) * ty;
    P[idx] = (bf16_t)v;
}

// ---------------------------------------------------------------------------
// bf16 WMMA GEMM:  D[M,N] = A[M,K] * B[N,K]^T (+ bias[n]), K % 32 == 0.
// Block: 256 thr = 8 waves, tile 128x128, BK=32, double-buffered LDS
// [2][128][40] bf16. Wave (wm 0..3, wn 0..1) owns 32x64 => 2x4 v8f accums.
// TDM path: wave 0 DMAs next tile pair while all waves compute; one barrier
// per K-step (s_wait_tensorcnt 0 -> barrier publishes the buffer; the DMA
// overwriting buffer b is issued only after the barrier proving everyone
// finished reading b).
// MODE 0: Fout bf16 = acc + bias           (tower)
// MODE 1: Out f32   = wgt * (acc + bias)   (projector, first expert: write)
// MODE 2: Out f32  += wgt * (acc + bias)   (projector, accumulate)
// ---------------------------------------------------------------------------
template<int MODE>
__global__ __launch_bounds__(256)
void gemm_bf16_wmma_kernel(const bf16_t* __restrict__ Ag, const bf16_t* __restrict__ Bg,
                           const float* __restrict__ bias, int K, int N,
                           bf16_t* __restrict__ Fout, float* __restrict__ Out,
                           const int* __restrict__ sel, const float* __restrict__ rw,
                           int expert)
{
    __shared__ __align__(16) bf16_t Al[2][128 * 40];
    __shared__ __align__(16) bf16_t Bl[2][128 * 40];

    const int tid  = threadIdx.x;
    const int lane = tid & 31;
    const int wv   = tid >> 5;
    const int wm   = wv & 3;                 // wave row (4 x 32 rows)
    const int wn   = wv >> 2;                // wave col (2 x 64 cols)
    const long blockM = (long)blockIdx.x * 128;
    const int  blockN = (int)blockIdx.y * 128;

    v8f acc[2][4];
#pragma unroll
    for (int mi = 0; mi < 2; ++mi)
#pragma unroll
        for (int ni = 0; ni < 4; ++ni)
#pragma unroll
            for (int r = 0; r < 8; ++r) acc[mi][ni][r] = 0.0f;

    const bf16_t* Ab = Ag + blockM * K;
    const bf16_t* Bb = Bg + (long)blockN * K;
    const int nK = K >> 5;

#if HAVE_TDM
    if (wv == 0) {                            // pre-issue tile 0 (TENSORcnt)
        tdm_load_tile(Ab, &Al[0][0], K);
        tdm_load_tile(Bb, &Bl[0][0], K);
    }
#endif

    for (int kt = 0; kt < nK; ++kt) {
        const int buf = kt & 1;
#if HAVE_TDM
        __builtin_amdgcn_s_wait_tensorcnt(0); // no-op for non-issuing waves
        __syncthreads();                      // publish buf to all waves
        if (kt + 1 < nK && wv == 0) {         // DMA next tile into buf^1
            const int k1 = (kt + 1) << 5;
            tdm_load_tile(Ab + k1, &Al[buf ^ 1][0], K);
            tdm_load_tile(Bb + k1, &Bl[buf ^ 1][0], K);
        }
#else
        const int k0 = kt << 5;
        // Cooperative tile load: 128 rows x 32 k = 512 x 16B chunks, 2/thread.
#pragma unroll
        for (int i = 0; i < 2; ++i) {
            int ch  = tid + (i << 8);
            int row = ch >> 2;
            int seg = (ch & 3) << 3;
            const bf16_t* ga = Ab + (long)row * K + k0 + seg;
            const bf16_t* gb = Bb + (long)row * K + k0 + seg;
            if (kt + 1 < nK) {                // -> global_prefetch_b8
                __builtin_prefetch(ga + 32, 0, 1);
                __builtin_prefetch(gb + 32, 0, 1);
            }
            *(v8bf*)&Al[buf][row * 40 + seg] = *(const v8bf*)ga;
            *(v8bf*)&Bl[buf][row * 40 + seg] = *(const v8bf*)gb;
        }
        __syncthreads();
#endif
        const bf16_t* As = &Al[buf][0];
        const bf16_t* Bs = &Bl[buf][0];

        // ISA bf16 16x32 A layout: lane L -> row L&15, kbase = (L>>4)*8;
        // elems 0..7 = K[kbase..+7], elems 8..15 = K[kbase+16..+23].
        const int kb   = (lane >> 4) << 3;
        const int rsel = lane & 15;
        v16bf afr[2], bfr[4];
#pragma unroll
        for (int mi = 0; mi < 2; ++mi) {
            const bf16_t* pa = &As[(wm * 32 + mi * 16 + rsel) * 40 + kb];
            v8bf lo = *(const v8bf*)pa;
            v8bf hi = *(const v8bf*)(pa + 16);
            afr[mi] = __builtin_shufflevector(lo, hi,
                0,1,2,3,4,5,6,7,8,9,10,11,12,13,14,15);
        }
#pragma unroll
        for (int ni = 0; ni < 4; ++ni) {
            const bf16_t* pb = &Bs[(wn * 64 + ni * 16 + rsel) * 40 + kb];
            v8bf lo = *(const v8bf*)pb;
            v8bf hi = *(const v8bf*)(pb + 16);
            bfr[ni] = __builtin_shufflevector(lo, hi,
                0,1,2,3,4,5,6,7,8,9,10,11,12,13,14,15);
        }
#pragma unroll
        for (int mi = 0; mi < 2; ++mi)
#pragma unroll
            for (int ni = 0; ni < 4; ++ni)
                acc[mi][ni] = __builtin_amdgcn_wmma_f32_16x16x32_bf16(
                    false, afr[mi], false, bfr[ni],
                    (short)0, acc[mi][ni], false, false);
#if !HAVE_TDM
        __syncthreads();                      // before overwriting buf next use
#endif
    }

    // Epilogue. 16x16 f32 D layout: lane L -> col L&15, rows (L>>4)*8 + r.
    float wgt = 1.0f;
    if (MODE != 0) {
        int b = (int)(blockM >> 10);          // 1024 proj rows/batch; 128|1024
        float w0 = (sel[2 * b + 0] == expert) ? rw[2 * b + 0] : 0.0f;
        float w1 = (sel[2 * b + 1] == expert) ? rw[2 * b + 1] : 0.0f;
        wgt = w0 + w1;
    }
    const int nsel  = lane & 15;
    const int mhalf = (lane >> 4) << 3;
#pragma unroll
    for (int mi = 0; mi < 2; ++mi) {
        const long mbase = blockM + wm * 32 + mi * 16 + mhalf;
#pragma unroll
        for (int ni = 0; ni < 4; ++ni) {
            const int n  = blockN + wn * 64 + ni * 16 + nsel;
            const float bn = bias[n];
#pragma unroll
            for (int r = 0; r < 8; ++r) {
                const float v = acc[mi][ni][r] + bn;
                const long  m = mbase + r;
                if (MODE == 0)      Fout[m * (long)N + n] = (bf16_t)v;
                else if (MODE == 1) Out[m * 1024 + n]  = wgt * v;
                else                Out[m * 1024 + n] += wgt * v;
            }
        }
    }
}

// ---------------------------------------------------------------------------
// Workspace arena (lifetime-aliased):
//   [0, 150994944)   : A (tower input) then P (resized feats) — disjoint lives
//   [+, +134217728)  : F (tower output; == P for expert 0, g=32)
//   [+, +8388608)    : WT (tower weights, transposed bf16)
//   [+, +4194304)    : WP (proj weights, transposed bf16)     total ~298 MB
// ---------------------------------------------------------------------------
static const size_t OFF_F  = 150994944;              // 65536*1152*2
static const size_t OFF_WT = OFF_F + 134217728;      // + 65536*1024*2
static const size_t OFF_WP = OFF_WT + 8388608;

extern "C" void kernel_launch(void* const* d_in, const int* in_sizes, int n_in,
                              void* d_out, int out_size, void* d_ws, size_t ws_size,
                              hipStream_t stream)
{
    (void)in_sizes; (void)n_in; (void)out_size; (void)ws_size;

    const float* xin = (const float*)d_in[0];
    const int*   sel = (const int*)d_in[1];
    const float* rw  = (const float*)d_in[2];
    float* out = (float*)d_out;

    char* ws = (char*)d_ws;
    bf16_t* A  = (bf16_t*)ws;                 // also P after tower GEMM runs
    bf16_t* F  = (bf16_t*)(ws + OFF_F);
    bf16_t* WT = (bf16_t*)(ws + OFF_WT);
    bf16_t* WP = (bf16_t*)(ws + OFF_WP);

    const int S_[3]  = {448, 336, 448};
    const int P_[3]  = {14, 14, 32};
    const int C_[3]  = {1024, 768, 1152};
    const int G_[3]  = {32, 24, 14};
    const int KP_[3] = {608, 608, 3072};      // 3*p*p padded to mult of 32

    for (int i = 0; i < 3; ++i) {
        const float* wt = (const float*)d_in[3 + 4 * i];
        const float* bt = (const float*)d_in[4 + 4 * i];
        const float* wp = (const float*)d_in[5 + 4 * i];
        const float* bp = (const float*)d_in[6 + 4 * i];
        const int s = S_[i], p = P_[i], c = C_[i], g = G_[i], kp = KP_[i];
        const int rows_t = 64 * g * g;

        // Weights -> transposed bf16 [N, Kpad]
        long nWT = (long)c * kp;
        convert_wT_kernel<<<(unsigned)((nWT + TB - 1) / TB), TB, 0, stream>>>(
            wt, WT, 3 * p * p, c, kp, nWT);
        long nWP = 1024L * c;                 // proj K == c, already 32-aligned
        convert_wT_kernel<<<(unsigned)((nWP + TB - 1) / TB), TB, 0, stream>>>(
            wp, WP, c, 1024, c, nWP);

        // Patch pack (+fused input resize for expert 1)
        long nA = (long)rows_t * kp;
        pack_patches_kernel<<<(unsigned)((nA + TB - 1) / TB), TB, 0, stream>>>(
            xin, A, g, p, kp, s, nA);

        // Tower GEMM: [rows_t, kp] x [c, kp]^T -> F bf16 (+ tower bias)
        dim3 gt((unsigned)(rows_t / 128), (unsigned)(c / 128));
        gemm_bf16_wmma_kernel<0><<<gt, TB, 0, stream>>>(
            A, WT, bt, kp, c, F, nullptr, nullptr, nullptr, i);

        // Feature-grid resize g -> 32 (identity for expert 0)
        const bf16_t* Pm;
        if (i == 0) {
            Pm = F;                           // g == 32 already
        } else {
            long nP = 64L * 1024 * c;         // A is dead; reuse its region
            resize_feat_kernel<<<(unsigned)((nP + TB - 1) / TB), TB, 0, stream>>>(
                F, A, g, c, nP);
            Pm = A;
        }

        // Projector GEMM: [65536, c] x [1024, c]^T -> out (routed, + bias)
        dim3 gq(512u, 8u);
        if (i == 0)
            gemm_bf16_wmma_kernel<1><<<gq, TB, 0, stream>>>(
                Pm, WP, bp, c, 1024, nullptr, out, sel, rw, i);
        else
            gemm_bf16_wmma_kernel<2><<<gq, TB, 0, stream>>>(
                Pm, WP, bp, c, 1024, nullptr, out, sel, rw, i);
    }
}
